// Standalone_HSTU_Model_64476049048227
// MI455X (gfx1250) — matrix operand order repaired and checked by reading, not compiled, not run on hardware
//
#include <hip/hip_runtime.h>
#include <hip/hip_bf16.h>

typedef _Float16 half_t;
typedef __attribute__((ext_vector_type(16))) _Float16 v16h;
typedef __attribute__((ext_vector_type(8)))  _Float16 v8h;
typedef __attribute__((ext_vector_type(8)))  float    v8f;

#define S_LEN 1024
#define B_SZ  4
#define E_DIM 256
#define H_CNT 4
#define D_HEAD 64
#define M_ROWS (B_SZ * S_LEN)      // 4096
#define NVOCAB 32001
#define NVOCAB_PAD 32064           // multiple of 64

#if __has_builtin(__builtin_amdgcn_global_load_async_to_lds_b128) && \
    __has_builtin(__builtin_amdgcn_s_wait_asynccnt)
#define HAS_ASYNC_LDS 1
#else
#define HAS_ASYNC_LDS 0
#endif

typedef int v4i __attribute__((vector_size(16)));
typedef __attribute__((address_space(1))) v4i as1_v4i;
typedef __attribute__((address_space(3))) v4i as3_v4i;

__device__ __forceinline__ v8f zero_v8f() {
    v8f z = {0.f, 0.f, 0.f, 0.f, 0.f, 0.f, 0.f, 0.f};
    return z;
}

__device__ __forceinline__ v16h cat8(v8h lo, v8h hi) {
    return __builtin_shufflevector(lo, hi, 0,1,2,3,4,5,6,7,8,9,10,11,12,13,14,15);
}

// A fragment, 16x32 (MxK), A row-major with leading dim lda (halves).
// Lane L<16: m=L, halves 0..7 = K 0..7,  halves 8..15 = K 16..23
// Lane L>=16: m=L-16, halves 0..7 = K 8..15, halves 8..15 = K 24..31
__device__ __forceinline__ v16h load_a_frag(const half_t* A, int lda) {
    int lane = threadIdx.x & 31;
    int m  = lane & 15;
    int kb = (lane >> 4) << 3;          // 0 or 8
    const half_t* p = A + m * lda + kb;
    v8h lo = *(const v8h*)(p);
    v8h hi = *(const v8h*)(p + 16);
    return cat8(lo, hi);
}

// B fragment, 32x16 (KxN), from row-major source W (N x K): B[k][n] = W[n][k].
// Lane L: n=L&15, kb = 8*(L>>4): halves 0..7 = B[kb+j][n], halves 8..15 = B[16+kb+j][n] (matches A frag).
__device__ __forceinline__ v16h load_b_frag_w(const half_t* W, int ldw) {
    int lane = threadIdx.x & 31;
    int n  = lane & 15;
    int kb = (lane >> 4) << 3;          // 0 or 8
    const half_t* p = W + n * ldw + kb;
    v8h lo = *(const v8h*)(p);
    v8h hi = *(const v8h*)(p + 16);
    return cat8(lo, hi);
}

__device__ __forceinline__ v8f wmma16(v16h a, v16h b, v8f c) {
    return __builtin_amdgcn_wmma_f32_16x16x32_f16(false, a, false, b,
                                                  (short)0, c, false, false);
}

__device__ __forceinline__ float silu_f(float x) {
    // fast rcp (single v_rcp_f32), not the IEEE divide expansion
    return x * __builtin_amdgcn_rcpf(1.f + __expf(-x));
}

// 16B global -> LDS copy, async (ASYNCcnt) path if the toolchain has it
__device__ __forceinline__ void copy16_g2l(const half_t* g, half_t* l) {
#if HAS_ASYNC_LDS
    __builtin_amdgcn_global_load_async_to_lds_b128((as1_v4i*)g, (as3_v4i*)l, 0, 0);
#else
    *(v8h*)l = *(const v8h*)g;
#endif
}

// ---------------- embedding + positional ----------------
__global__ void embed_kernel(const int* __restrict__ seq,
                             const float* __restrict__ emb,
                             const float* __restrict__ pos,
                             float* __restrict__ X) {
    int e  = threadIdx.x;               // 0..255
    int bs = blockIdx.x;                // 0..4095 == b*S + s
    int s  = bs & (S_LEN - 1);
    int tok = seq[bs];
    X[(size_t)bs * E_DIM + e] = emb[(size_t)tok * E_DIM + e] + pos[s * E_DIM + e];
}

// ---------------- layernorm (one wave per row) -> f16 ----------------
__global__ void ln_kernel(const float* __restrict__ X,
                          const float* __restrict__ g,
                          const float* __restrict__ bb,
                          half_t* __restrict__ Out, int rows) {
    int lane = threadIdx.x & 31;
    int wid  = threadIdx.x >> 5;
    int row  = blockIdx.x * (blockDim.x >> 5) + wid;
    if (row >= rows) return;
    const float* x = X + (size_t)row * E_DIM;
    float vals[8];
    float s = 0.f;
#pragma unroll
    for (int i = 0; i < 8; ++i) { vals[i] = x[lane + 32 * i]; s += vals[i]; }
#pragma unroll
    for (int off = 16; off > 0; off >>= 1) s += __shfl_xor(s, off, 32);
    float mean = s * (1.f / 256.f);
    float v = 0.f;
#pragma unroll
    for (int i = 0; i < 8; ++i) { float d = vals[i] - mean; v += d * d; }
#pragma unroll
    for (int off = 16; off > 0; off >>= 1) v += __shfl_xor(v, off, 32);
    float inv = __frsqrt_rn(v * (1.f / 256.f) + 1e-5f);
#pragma unroll
    for (int i = 0; i < 8; ++i) {
        int c = lane + 32 * i;
        Out[(size_t)row * E_DIM + c] =
            (half_t)((vals[i] - mean) * inv * g[c] + bb[c]);
    }
}

// ---------------- weight conversion ----------------
__global__ void f32_to_f16_kernel(const float* __restrict__ in,
                                  half_t* __restrict__ out, int n) {
    int i = blockIdx.x * blockDim.x + threadIdx.x;
    if (i < n) out[i] = (half_t)in[i];
}

__global__ void head_pad_kernel(const float* __restrict__ W,
                                half_t* __restrict__ out) {
    size_t i = (size_t)blockIdx.x * blockDim.x + threadIdx.x;
    size_t total = (size_t)NVOCAB_PAD * E_DIM;
    if (i >= total) return;
    size_t row = i / E_DIM;
    out[i] = (row < NVOCAB) ? (half_t)W[i] : (half_t)0.f;
}

// ---------------- generic wave-tiled GEMM: C = A(MxK) * W(NxK)^T + bias ----
// Block = 8 waves = 128(M) x 64(N) tile; K = 256 fixed.
// W tile (64 x 256, 32 KB) is staged in LDS once per block (async if available).
// MODE 0: OutH[m*ldo+n] = f16(silu(acc+bias))
// MODE 1: Xres[m*ldo+n] += acc+bias            (residual, fp32)
// MODE 2: Xres[m*ldo+n]  = acc+bias, n < Nvalid (logits, fp32)
template <int MODE>
__global__ __launch_bounds__(256)
void gemm_wave_kernel(const half_t* __restrict__ A,
                      const half_t* __restrict__ W,
                      const float* __restrict__ bias,
                      float* __restrict__ Xres,
                      half_t* __restrict__ OutH,
                      int Nvalid, int ldo) {
    __shared__ __align__(16) half_t wtile[64 * 256];   // 32 KB

    int wave  = threadIdx.x >> 5;
    int mBase = blockIdx.y * 128 + wave * 16;
    int nBase = blockIdx.x * 64;

    {   // cooperative stage of W rows [nBase, nBase+64) x K=256 into LDS
        int row = threadIdx.x >> 2;          // 0..63
        int seg = (threadIdx.x & 3) * 64;    // 0,64,128,192
        const half_t* src = W + (size_t)(nBase + row) * 256 + seg;
        half_t* dst = wtile + row * 256 + seg;
#pragma unroll
        for (int j = 0; j < 8; ++j)
            copy16_g2l(src + 8 * j, dst + 8 * j);
    }
#if HAS_ASYNC_LDS
    __builtin_amdgcn_s_wait_asynccnt(0);
#endif
    __syncthreads();

    v8f acc[4] = {zero_v8f(), zero_v8f(), zero_v8f(), zero_v8f()};
#pragma unroll
    for (int k0 = 0; k0 < 256; k0 += 32) {
        v16h a = load_a_frag(A + (size_t)mBase * 256 + k0, 256);
#pragma unroll
        for (int c = 0; c < 4; ++c) {
            v16h b = load_b_frag_w(wtile + (size_t)(16 * c) * 256 + k0, 256);
            acc[c] = wmma16(a, b, acc[c]);
        }
    }

    int lane = threadIdx.x & 31;
    int mOff = (lane >> 4) * 8;
    int nOff = lane & 15;
#pragma unroll
    for (int c = 0; c < 4; ++c) {
        int n = nBase + 16 * c + nOff;
        float bv = (MODE == 2 && n >= Nvalid) ? 0.f : bias[n];
#pragma unroll
        for (int r = 0; r < 8; ++r) {
            int m = mBase + mOff + r;
            float v = acc[c][r] + bv;
            if (MODE == 0) {
                OutH[(size_t)m * ldo + n] = (half_t)silu_f(v);
            } else if (MODE == 1) {
                Xres[(size_t)m * ldo + n] += v;
            } else {
                if (n < Nvalid) Xres[(size_t)m * ldo + n] = v;
            }
        }
    }
}

// ---------------- causal SiLU attention, one wave per (b, h, 16-row block) --
// P16: (B*S) x 512 f16; q = cols [h*64, h*64+64), u = cols [256+h*64, ...)
// OU : (B*S) x 256 f16, OU = o * u
// Per 32-wide t-step:
//   * lane t stages its q row into a TRANSPOSED LDS tile qtT[d][t] (b16
//     scatter stores; transpose paid on the store side, no load waits)
//   * score B-frags: contiguous b128 loads straight from global (L2 hits)
//   * silu(score) tile re-fragmented via LDS as A operand
//   * o B-frags: contiguous b128 LDS loads from qtT
// Single wave per block -> DS in-order, no barriers needed.
#define QTT_LD 40   // padded leading dim (halves): rows 16B-aligned, bank-spread
__global__ __launch_bounds__(32)
void attn_kernel(const half_t* __restrict__ P16, half_t* __restrict__ OU) {
    __shared__ __align__(16) half_t qtT[D_HEAD * QTT_LD];  // 5 KB, [d][t]
    __shared__ __align__(16) half_t at[16 * 32];           // silu(score) tile

    int sblock = blockIdx.x * 16;
    int h = blockIdx.y;
    int b = blockIdx.z;
    const half_t* Q = P16 + (size_t)(b * S_LEN) * 512 + h * D_HEAD;

    int lane = threadIdx.x & 31;
    int mOff = (lane >> 4) * 8;
    int nOff = lane & 15;

    // A fragments of q rows (constant over t loop): K = d = 0..63
    v16h aq0 = load_a_frag(Q + (size_t)sblock * 512 + 0, 512);
    v16h aq1 = load_a_frag(Q + (size_t)sblock * 512 + 32, 512);

    v8f oacc[4] = {zero_v8f(), zero_v8f(), zero_v8f(), zero_v8f()};

    for (int t0 = 0; t0 <= sblock; t0 += 32) {
        // stage q rows [t0, t0+32) transposed: lane t scatters its row
        {
            const half_t* src = Q + (size_t)(t0 + lane) * 512;
            v8h rv[8];
#pragma unroll
            for (int j = 0; j < 8; ++j) rv[j] = *(const v8h*)(src + 8 * j);
#pragma unroll
            for (int j = 0; j < 8; ++j)
#pragma unroll
                for (int e = 0; e < 8; ++e)
                    qtT[(8 * j + e) * QTT_LD + lane] = rv[j][e];
        }

#pragma unroll
        for (int ht = 0; ht < 2; ++ht) {
            int tb = t0 + ht * 16;
            if (tb <= sblock + 15) {
                v8f sacc = zero_v8f();
                sacc = wmma16(aq0, load_b_frag_w(Q + (size_t)tb * 512 + 0, 512), sacc);
                sacc = wmma16(aq1, load_b_frag_w(Q + (size_t)tb * 512 + 32, 512), sacc);
#pragma unroll
                for (int r = 0; r < 8; ++r) {
                    int sg = sblock + mOff + r;
                    int tg = tb + nOff;
                    float v = sacc[r] * 0.125f;   // 1/sqrt(64)
                    v = (tg <= sg) ? silu_f(v) : 0.f;
                    at[(mOff + r) * 32 + ht * 16 + nOff] = (half_t)v;
                }
            } else {
#pragma unroll
                for (int r = 0; r < 8; ++r)
                    at[(mOff + r) * 32 + ht * 16 + nOff] = (half_t)0.f;
            }
        }
        // Re-fragment attn tile (16x32) from LDS as A operand
        v16h aA = load_a_frag(at, 32);
        // o B-frags: B[k=t][n=d] = qtT[d][t], contiguous along t per lane
#pragma unroll
        for (int c = 0; c < 4; ++c)
            oacc[c] = wmma16(aA, load_b_frag_w(qtT + (size_t)(c * 16) * QTT_LD, QTT_LD),
                             oacc[c]);
    }

    // fused o * u -> f16
#pragma unroll
    for (int c = 0; c < 4; ++c) {
#pragma unroll
        for (int r = 0; r < 8; ++r) {
            int m = sblock + mOff + r;
            int d = c * 16 + nOff;
            size_t row = (size_t)b * S_LEN + m;
            float u = (float)P16[row * 512 + 256 + h * D_HEAD + d];
            OU[row * E_DIM + h * D_HEAD + d] = (half_t)(oacc[c][r] * u);
        }
    }
}

extern "C" void kernel_launch(void* const* d_in, const int* in_sizes, int n_in,
                              void* d_out, int out_size, void* d_ws, size_t ws_size,
                              hipStream_t stream) {
    const int*   seq    = (const int*)  d_in[0];
    const float* emb    = (const float*)d_in[1];
    const float* pos    = (const float*)d_in[2];
    const float* ln1_g  = (const float*)d_in[3];
    const float* ln1_b  = (const float*)d_in[4];
    const float* W_in   = (const float*)d_in[5];
    const float* b_in   = (const float*)d_in[6];
    const float* W_out  = (const float*)d_in[7];
    const float* b_out  = (const float*)d_in[8];
    const float* lnf_g  = (const float*)d_in[9];
    const float* lnf_b  = (const float*)d_in[10];
    const float* W_head = (const float*)d_in[11];
    const float* b_head = (const float*)d_in[12];
    float* logits = (float*)d_out;

    // workspace carve-up (all 256B aligned)
    char* ws = (char*)d_ws;
    size_t off = 0;
    auto carve = [&](size_t bytes) {
        void* p = ws + off;
        off += (bytes + 255) & ~(size_t)255;
        return p;
    };
    float*  x       = (float*) carve((size_t)M_ROWS * E_DIM * 4);      // 4 MB
    half_t* p16     = (half_t*)carve((size_t)M_ROWS * 512 * 2);        // 4 MB
    half_t* xn16    = (half_t*)carve((size_t)M_ROWS * E_DIM * 2);      // 2 MB
    half_t* ou16    = (half_t*)carve((size_t)M_ROWS * E_DIM * 2);      // 2 MB
    half_t* Win16   = (half_t*)carve((size_t)512 * E_DIM * 2);
    half_t* Wout16  = (half_t*)carve((size_t)E_DIM * E_DIM * 2);
    half_t* Whead16 = (half_t*)carve((size_t)NVOCAB_PAD * E_DIM * 2);  // 16.4 MB

    // weight conversions (fp32 -> f16, head padded to 32064 rows)
    f32_to_f16_kernel<<<(512 * E_DIM + 255) / 256, 256, 0, stream>>>(W_in, Win16, 512 * E_DIM);
    f32_to_f16_kernel<<<(E_DIM * E_DIM + 255) / 256, 256, 0, stream>>>(W_out, Wout16, E_DIM * E_DIM);
    head_pad_kernel<<<((NVOCAB_PAD * E_DIM) + 255) / 256, 256, 0, stream>>>(W_head, Whead16);

    // x = emb[seq] + pos
    embed_kernel<<<M_ROWS, E_DIM, 0, stream>>>(seq, emb, pos, x);

    for (int layer = 0; layer < 2; ++layer) {
        // xn = LN(x) -> f16
        ln_kernel<<<M_ROWS / 8, 256, 0, stream>>>(x, ln1_g, ln1_b, xn16, M_ROWS);
        // p = silu(xn @ W_in^T + b_in)  (4096 x 512)
        gemm_wave_kernel<0><<<dim3(512 / 64, M_ROWS / 128), 256, 0, stream>>>(
            xn16, Win16, b_in, nullptr, p16, 512, 512);
        // causal silu attention + fused o*u
        attn_kernel<<<dim3(S_LEN / 16, H_CNT, B_SZ), 32, 0, stream>>>(p16, ou16);
        // x += (o*u) @ W_out^T + b_out
        gemm_wave_kernel<1><<<dim3(E_DIM / 64, M_ROWS / 128), 256, 0, stream>>>(
            ou16, Wout16, b_out, x, nullptr, E_DIM, E_DIM);
    }

    // final LN + head
    ln_kernel<<<M_ROWS / 8, 256, 0, stream>>>(x, lnf_g, lnf_b, xn16, M_ROWS);
    gemm_wave_kernel<2><<<dim3(NVOCAB_PAD / 64, M_ROWS / 128), 256, 0, stream>>>(
        xn16, Whead16, b_head, logits, nullptr, NVOCAB, NVOCAB);
}
